// AdaptiveMCTSReasoner_37864431681835
// MI455X (gfx1250) — matrix operand order repaired and checked by reading, not compiled, not run on hardware
//
#include <hip/hip_runtime.h>
#include <math.h>

#define B_ 16
#define S_ 1024
#define H_ 1024
#define MAXSIMS 100

typedef __bf16 bf16;
typedef __attribute__((ext_vector_type(4)))  __bf16 v4bf;
typedef __attribute__((ext_vector_type(8)))  __bf16 v8bf;
typedef __attribute__((ext_vector_type(16))) __bf16 v16bf;
typedef __attribute__((ext_vector_type(8)))  float  v8f;

__device__ __forceinline__ float gelu_exact(float x) {
    return 0.5f * x * (1.0f + erff(x * 0.70710678118654752f));
}

__device__ __forceinline__ v4bf cvt4(float4 f) {
    v4bf r; r[0] = (bf16)f.x; r[1] = (bf16)f.y; r[2] = (bf16)f.z; r[3] = (bf16)f.w;
    return r;
}

// ---------------------------------------------------------------- utilities
__global__ void zero_f32(float* __restrict__ p, int n) {
    int i = blockIdx.x * blockDim.x + threadIdx.x;
    if (i < n) p[i] = 0.0f;
}

__global__ void root_mean_kernel(const float* __restrict__ root, float* __restrict__ rmean) {
    int h = blockIdx.x * blockDim.x + threadIdx.x;
    int b = blockIdx.y;
    const float* base = root + (size_t)b * S_ * H_ + h;
    float s = 0.f;
    for (int i = 0; i < S_; ++i) s += base[(size_t)i * H_];
    rmean[b * H_ + h] = s * (1.0f / S_);
}

// ------------------------------------------------- controller: sims[b] (tiny)
__global__ __launch_bounds__(256) void controller_kernel(
    const float* __restrict__ hidden, const float* __restrict__ W1, const float* __restrict__ b1,
    const float* __restrict__ W2, const float* __restrict__ b2, int* __restrict__ sims)
{
    __shared__ float x0[H_];
    __shared__ float h1[H_];
    __shared__ float lg[8];
    int b = blockIdx.x, tid = threadIdx.x;
    const float* src = hidden + (size_t)b * S_ * H_;   // row s=0
    for (int i = 0; i < 4; ++i) x0[tid + i * 256] = src[tid + i * 256];
    __syncthreads();
    float a0 = b1[tid], a1 = b1[tid + 256], a2 = b1[tid + 512], a3 = b1[tid + 768];
    for (int k = 0; k < H_; ++k) {
        float xv = x0[k];
        const float* wr = W1 + (size_t)k * H_ + tid;
        a0 += xv * wr[0]; a1 += xv * wr[256]; a2 += xv * wr[512]; a3 += xv * wr[768];
    }
    h1[tid] = gelu_exact(a0); h1[tid + 256] = gelu_exact(a1);
    h1[tid + 512] = gelu_exact(a2); h1[tid + 768] = gelu_exact(a3);
    __syncthreads();
    if (tid < 5) {
        float a = b2[tid];
        for (int k = 0; k < H_; ++k) a += h1[k] * W2[k * 5 + tid];
        lg[tid] = a;
    }
    __syncthreads();
    if (tid == 0) {
        const int opts[5] = {10, 25, 50, 75, 100};
        int am = 0; float bv = lg[0];
        for (int j = 1; j < 5; ++j) if (lg[j] > bv) { bv = lg[j]; am = j; }
        sims[b] = opts[am];
    }
}

// -------------------------------------------------------- WMMA bf16 GEMM
// C[M,N] = epi(concat(A1,A2)[M,K] @ W[K,N] + bias [+ residual])
// A1 row-stride = K1, A2 row-stride = K-K1.  EPI: 0 none, 1 GELU, 2 +residual
// Block tile 128x128, BK=32, 8 waves, each wave 64x32 (4x2 WMMA tiles).
// Double-buffered LDS; fragments loaded as aligned b128 pairs.
#define BM 128
#define BN 128
#define BK 32

template <int EPI>
__global__ __launch_bounds__(256) void gemm_wmma_bf16(
    const float* __restrict__ A1, const float* __restrict__ A2, int K1,
    const float* __restrict__ W, const float* __restrict__ bias,
    const float* __restrict__ residual, float* __restrict__ C,
    int M, int N, int K)
{
    __shared__ __align__(16) bf16 As[2][BM][BK];   // 2 x 8KB, row = 64B
    __shared__ __align__(16) bf16 Bt[2][BN][BK];   // 2 x 8KB, transposed [n][k]
    int tid = threadIdx.x;
    int lane = tid & 31, wave = tid >> 5;
    int wm = wave & 1;          // 2 wave-rows of 64
    int wn = wave >> 1;         // 4 wave-cols of 32
    int rowBase = blockIdx.y * BM;
    int colBase = blockIdx.x * BN;
    int ln = lane & 15;
    int hi = lane >> 4;
    int ka = hi ? 8 : 0;        // A-frag K base (16-bit A 16x32 layout)
    int kb = hi ? 16 : 0;       // unused; B frag also uses ka (same k order as A)

    v8f acc[4][2] = {};

    float4 ar[4], br[4];

    auto loadA = [&](int k0) {
        #pragma unroll
        for (int i = 0; i < 4; ++i) {
            int idx = i * 256 + tid;           // 1024 float4 tasks
            int r = idx >> 3, c4 = idx & 7;    // 8 float4 per row
            int col = k0 + c4 * 4;
            const float* src = (col < K1)
                ? (A1 + (size_t)(rowBase + r) * K1 + col)
                : (A2 + (size_t)(rowBase + r) * (K - K1) + (col - K1));
            ar[i] = *(const float4*)src;
        }
    };
    auto loadB = [&](int k0) {
        #pragma unroll
        for (int i = 0; i < 4; ++i) {
            int task = i * 256 + tid;          // 1024 tasks: (n, k-group of 4)
            int n = task & 127, kg = task >> 7;
            const float* base = W + (size_t)(k0 + kg * 4) * N + colBase + n;
            float4 v;
            v.x = base[0]; v.y = base[N]; v.z = base[2 * N]; v.w = base[3 * N];
            br[i] = v;
        }
    };
    auto storeA = [&](int buf) {
        #pragma unroll
        for (int i = 0; i < 4; ++i) {
            int idx = i * 256 + tid;
            int r = idx >> 3, c4 = idx & 7;
            *(v4bf*)&As[buf][r][c4 * 4] = cvt4(ar[i]);
        }
    };
    auto storeB = [&](int buf) {
        #pragma unroll
        for (int i = 0; i < 4; ++i) {
            int task = i * 256 + tid;
            int n = task & 127, kg = task >> 7;
            *(v4bf*)&Bt[buf][n][kg * 4] = cvt4(br[i]);
        }
    };
    auto compute = [&](int buf) {
        v16bf bfrag[2];
        #pragma unroll
        for (int nt = 0; nt < 2; ++nt) {
            const v8bf* bp = (const v8bf*)&Bt[buf][wn * 32 + nt * 16 + ln][0];
            v8bf b0 = bp[ka >> 3];             // k = ka .. ka+7
            v8bf b1 = bp[2 + (ka >> 3)];       // k = 16+ka .. 23+ka
            bfrag[nt] = __builtin_shufflevector(b0, b1,
                0, 1, 2, 3, 4, 5, 6, 7, 8, 9, 10, 11, 12, 13, 14, 15);
        }
        #pragma unroll
        for (int mt = 0; mt < 4; ++mt) {
            const v8bf* ap = (const v8bf*)&As[buf][wm * 64 + mt * 16 + ln][0];
            v8bf a0 = ap[ka >> 3];             // k = ka .. ka+7
            v8bf a1 = ap[2 + (ka >> 3)];       // k = 16+ka .. 23+ka
            v16bf afrag = __builtin_shufflevector(a0, a1,
                0, 1, 2, 3, 4, 5, 6, 7, 8, 9, 10, 11, 12, 13, 14, 15);
            acc[mt][0] = __builtin_amdgcn_wmma_f32_16x16x32_bf16(
                false, afrag, false, bfrag[0], (short)0, acc[mt][0], false, false);
            acc[mt][1] = __builtin_amdgcn_wmma_f32_16x16x32_bf16(
                false, afrag, false, bfrag[1], (short)0, acc[mt][1], false, false);
        }
    };

    int nIter = K / BK;
    loadA(0); loadB(0);
    storeA(0); storeB(0);
    __syncthreads();
    for (int it = 0; it < nIter; ++it) {
        int cur = it & 1;
        bool more = (it + 1) < nIter;
        if (more) { loadA((it + 1) * BK); loadB((it + 1) * BK); }  // overlap with MMA
        compute(cur);
        if (more) { storeA(1 - cur); storeB(1 - cur); }
        __syncthreads();
    }

    #pragma unroll
    for (int nt = 0; nt < 2; ++nt) {
        int colC = colBase + wn * 32 + nt * 16 + ln;
        float bv = bias[colC];
        #pragma unroll
        for (int mt = 0; mt < 4; ++mt) {
            #pragma unroll
            for (int v = 0; v < 8; ++v) {
                int rowC = rowBase + wm * 64 + mt * 16 + v + (hi ? 8 : 0);
                size_t o = (size_t)rowC * N + colC;
                float x = acc[mt][nt][v] + bv;
                if (EPI == 1) x = gelu_exact(x);
                if (EPI == 2) x += residual[o];
                C[o] = x;
            }
        }
    }
}

// ------------------------------------- colmean(Wp2) and mean(bp2) for GEMV
__global__ void colmean_kernel(const float* __restrict__ Wp2, const float* __restrict__ bp2,
                               float* __restrict__ cm, float* __restrict__ mbp2) {
    int k = blockIdx.x * blockDim.x + threadIdx.x;
    if (k < H_) {
        float s = 0.f;
        const float* r = Wp2 + (size_t)k * H_;
        for (int j = 0; j < H_; ++j) s += r[j];
        cm[k] = s * (1.0f / H_);
    }
    if (blockIdx.x == 0 && threadIdx.x == 0) {
        float s = 0.f;
        for (int j = 0; j < H_; ++j) s += bp2[j];
        mbp2[0] = s * (1.0f / H_);
    }
}

// ------------------------------------ focus_logits[b,s] = h1 . cm  (+mask)
__global__ __launch_bounds__(256) void focus_logits_kernel(
    const float* __restrict__ h1, const float* __restrict__ cm, const float* __restrict__ mbp2,
    const int* __restrict__ amask, float* __restrict__ fl)
{
    int wave = threadIdx.x >> 5, lane = threadIdx.x & 31;
    int row = blockIdx.x * 8 + wave;
    const float* hr = h1 + (size_t)row * H_;
    float s = 0.f;
    for (int i = 0; i < H_ / 32; ++i) { int k = lane + i * 32; s += hr[k] * cm[k]; }
    for (int off = 16; off > 0; off >>= 1) s += __shfl_down(s, off, 32);
    if (lane == 0) {
        float v = s + mbp2[0];
        if (amask[row] == 0) v = -1e9f;
        fl[row] = v;
    }
}

// -------------------------------------------- Gumbel top-3 per (t,b)
__device__ __forceinline__ float gumbel_noise(int t, int b, int s) {
    unsigned x = 0x9E3779B9u ^ (unsigned)(t * 1000003) ^ (unsigned)(b * 7919) ^ (unsigned)(s * 104729);
    x ^= x >> 16; x *= 0x7FEB352Du; x ^= x >> 15; x *= 0x846CA68Bu; x ^= x >> 16;
    float u = (float)(x >> 8) * (1.0f / 16777216.0f);
    u = fminf(fmaxf(u, 1e-12f), 1.0f - 1e-7f);
    return -logf(-logf(u));
}

__global__ __launch_bounds__(256) void topk_kernel(const float* __restrict__ fl,
                                                   int* __restrict__ fpos) {
    __shared__ float tv[768];
    __shared__ int   ti[768];
    int b = blockIdx.x, t = blockIdx.y, tid = threadIdx.x;
    const float* row = fl + b * S_;
    float v0 = -INFINITY, v1 = -INFINITY, v2 = -INFINITY;
    int   i0 = 0, i1 = 0, i2 = 0;
    for (int i = 0; i < 4; ++i) {
        int s = tid + i * 256;
        float v = row[s] + gumbel_noise(t, b, s);
        if (v > v0)      { v2 = v1; i2 = i1; v1 = v0; i1 = i0; v0 = v; i0 = s; }
        else if (v > v1) { v2 = v1; i2 = i1; v1 = v;  i1 = s; }
        else if (v > v2) { v2 = v;  i2 = s; }
    }
    tv[tid] = v0; ti[tid] = i0;
    tv[256 + tid] = v1; ti[256 + tid] = i1;
    tv[512 + tid] = v2; ti[512 + tid] = i2;
    __syncthreads();
    if (tid == 0) {
        float w0 = -INFINITY, w1 = -INFINITY, w2 = -INFINITY;
        int   j0 = 0, j1 = 0, j2 = 0;
        for (int i = 0; i < 768; ++i) {
            float v = tv[i]; int ix = ti[i];
            if (v > w0)      { w2 = w1; j2 = j1; w1 = w0; j1 = j0; w0 = v; j0 = ix; }
            else if (v > w1) { w2 = w1; j2 = j1; w1 = v;  j1 = ix; }
            else if (v > w2) { w2 = v;  j2 = ix; }
        }
        int base = (t * B_ + b) * 3;
        fpos[base] = j0; fpos[base + 1] = j1; fpos[base + 2] = j2;
    }
}

// -------------------- one (t,b) simulation: 3 trans MLPs + value head,
// sparse delta scatter into accb, weight into wsum. Steps independent over t.
__global__ __launch_bounds__(256) void sim_kernel(
    const float* __restrict__ root, const float* __restrict__ rmean,
    const int* __restrict__ fpos, const int* __restrict__ sims,
    const float* __restrict__ Wt1, const float* __restrict__ bt1,
    const float* __restrict__ Wt2, const float* __restrict__ bt2,
    const float* __restrict__ Wav1, const float* __restrict__ bav1,
    const float* __restrict__ Wav2, const float* __restrict__ bav2,
    float* __restrict__ accb, float* __restrict__ wsum)
{
    __shared__ float sm_m[H_];        // running current.mean
    __shared__ float sm_x2[H_];       // focus state
    __shared__ float sm_h[H_];        // hidden activations
    __shared__ float sm_d[3][H_];     // per-j deltas (new - root[pos])
    __shared__ float red[256];
    __shared__ float sm_w;
    __shared__ int   sm_pos[3];
    int b = blockIdx.x, t = blockIdx.y, tid = threadIdx.x;
    const float* rm = rmean + b * H_;
    for (int i = 0; i < 4; ++i) sm_m[tid + i * 256] = rm[tid + i * 256];
    if (tid < 3) sm_pos[tid] = fpos[(t * B_ + b) * 3 + tid];
    __syncthreads();

    for (int j = 0; j < 3; ++j) {
        int pos = sm_pos[j];
        const float* fs = root + ((size_t)b * S_ + pos) * H_;
        for (int i = 0; i < 4; ++i) sm_x2[tid + i * 256] = fs[tid + i * 256];
        __syncthreads();
        // layer1: [2048] -> [1024], GELU
        float a0 = bt1[tid], a1 = bt1[tid + 256], a2 = bt1[tid + 512], a3 = bt1[tid + 768];
        for (int k = 0; k < 2 * H_; ++k) {
            float inv = (k < H_) ? sm_m[k] : sm_x2[k - H_];
            const float* wr = Wt1 + (size_t)k * H_ + tid;
            a0 += inv * wr[0]; a1 += inv * wr[256]; a2 += inv * wr[512]; a3 += inv * wr[768];
        }
        sm_h[tid] = gelu_exact(a0); sm_h[tid + 256] = gelu_exact(a1);
        sm_h[tid + 512] = gelu_exact(a2); sm_h[tid + 768] = gelu_exact(a3);
        __syncthreads();
        // layer2: [1024] -> [1024], delta vs root[pos]
        float c0 = bt2[tid], c1 = bt2[tid + 256], c2 = bt2[tid + 512], c3 = bt2[tid + 768];
        for (int k = 0; k < H_; ++k) {
            float hv = sm_h[k];
            const float* wr = Wt2 + (size_t)k * H_ + tid;
            c0 += hv * wr[0]; c1 += hv * wr[256]; c2 += hv * wr[512]; c3 += hv * wr[768];
        }
        sm_d[j][tid]       = c0 - sm_x2[tid];
        sm_d[j][tid + 256] = c1 - sm_x2[tid + 256];
        sm_d[j][tid + 512] = c2 - sm_x2[tid + 512];
        sm_d[j][tid + 768] = c3 - sm_x2[tid + 768];
        __syncthreads();
        for (int i = 0; i < 4; ++i) { int n = tid + i * 256; sm_m[n] += sm_d[j][n] * (1.0f / S_); }
        __syncthreads();
    }

    // value head: [root_mean | current.mean] -> H -> 1 -> sigmoid
    float a0 = bav1[tid], a1 = bav1[tid + 256], a2 = bav1[tid + 512], a3 = bav1[tid + 768];
    for (int k = 0; k < 2 * H_; ++k) {
        float inv = (k < H_) ? rm[k] : sm_m[k - H_];
        const float* wr = Wav1 + (size_t)k * H_ + tid;
        a0 += inv * wr[0]; a1 += inv * wr[256]; a2 += inv * wr[512]; a3 += inv * wr[768];
    }
    red[tid] = gelu_exact(a0) * Wav2[tid] + gelu_exact(a1) * Wav2[tid + 256] +
               gelu_exact(a2) * Wav2[tid + 512] + gelu_exact(a3) * Wav2[tid + 768];
    __syncthreads();
    for (int off = 128; off > 0; off >>= 1) {
        if (tid < off) red[tid] += red[tid + off];
        __syncthreads();
    }
    if (tid == 0) {
        float val = red[0] + bav2[0];
        float w = 1.0f / (1.0f + expf(-val));
        float wv = (t < sims[b]) ? w : 0.0f;
        sm_w = wv;
        atomicAdd(&wsum[b], wv);
    }
    __syncthreads();
    float wv = sm_w;
    if (wv != 0.0f) {
        for (int j = 0; j < 3; ++j) {
            float* dst = accb + ((size_t)b * S_ + sm_pos[j]) * H_;
            for (int i = 0; i < 4; ++i) {
                int n = tid + i * 256;
                atomicAdd(&dst[n], wv * sm_d[j][n]);
            }
        }
    }
}

// -------------------------------- acc = wsum[b]*root + sparse-delta sum
__global__ void acc_finalize(const float* __restrict__ root, const float* __restrict__ wsum,
                             float* __restrict__ accb) {
    size_t i = (size_t)blockIdx.x * blockDim.x + threadIdx.x;
    int b = (int)(i >> 20);                           // S_*H_ = 2^20
    accb[i] = wsum[b] * root[i] + accb[i];
}

// ---------------------------------------------------------------- launch
extern "C" void kernel_launch(void* const* d_in, const int* in_sizes, int n_in,
                              void* d_out, int out_size, void* d_ws, size_t ws_size,
                              hipStream_t stream) {
    (void)in_sizes; (void)n_in; (void)out_size; (void)ws_size;
    const float* hidden = (const float*)d_in[0];
    const int*   amask  = (const int*)d_in[1];
    const float* Wsc1 = (const float*)d_in[2];  const float* bsc1 = (const float*)d_in[3];
    const float* Wsc2 = (const float*)d_in[4];  const float* bsc2 = (const float*)d_in[5];
    const float* Wp1  = (const float*)d_in[6];  const float* bp1  = (const float*)d_in[7];
    const float* Wp2  = (const float*)d_in[8];  const float* bp2  = (const float*)d_in[9];
    const float* Wt1  = (const float*)d_in[10]; const float* bt1  = (const float*)d_in[11];
    const float* Wt2  = (const float*)d_in[12]; const float* bt2  = (const float*)d_in[13];
    const float* Wav1 = (const float*)d_in[14]; const float* bav1 = (const float*)d_in[15];
    const float* Wav2 = (const float*)d_in[16]; const float* bav2 = (const float*)d_in[17];
    const float* Wg1  = (const float*)d_in[18]; const float* bg1  = (const float*)d_in[19];
    const float* Wg2  = (const float*)d_in[20]; const float* bg2  = (const float*)d_in[21];
    float* out = (float*)d_out;

    const size_t BSH = (size_t)B_ * S_ * H_;          // 16M floats
    float* ws    = (float*)d_ws;
    float* h1    = ws;                                // 16M
    float* accb  = h1 + BSH;                          // 16M
    float* fl    = accb + BSH;                        // 16384
    float* cm    = fl + B_ * S_;                      // 1024
    float* mbp2  = cm + H_;                           // 1 (padded 64)
    float* rmean = mbp2 + 64;                         // 16*1024
    float* wsum  = rmean + B_ * H_;                   // 16 (padded 64)
    int*   sims  = (int*)(wsum + 64);                 // 16 (padded 64)
    int*   fpos  = sims + 64;                         // 100*16*3

    zero_f32<<<(int)(BSH / 256), 256, 0, stream>>>(accb, (int)BSH);
    zero_f32<<<1, 64, 0, stream>>>(wsum, 16);

    root_mean_kernel<<<dim3(H_ / 256, B_), 256, 0, stream>>>(hidden, rmean);
    controller_kernel<<<B_, 256, 0, stream>>>(hidden, Wsc1, bsc1, Wsc2, bsc2, sims);

    // policy layer 1 (WMMA): h1 = GELU(root @ Wp1 + bp1)
    gemm_wmma_bf16<1><<<dim3(H_ / BN, (B_ * S_) / BM), 256, 0, stream>>>(
        hidden, nullptr, H_, Wp1, bp1, nullptr, h1, B_ * S_, H_, H_);

    // policy layer 2 collapsed to GEMV via column-mean of Wp2
    colmean_kernel<<<H_ / 256, 256, 0, stream>>>(Wp2, bp2, cm, mbp2);
    focus_logits_kernel<<<(B_ * S_) / 8, 256, 0, stream>>>(h1, cm, mbp2, amask, fl);

    topk_kernel<<<dim3(B_, MAXSIMS), 256, 0, stream>>>(fl, fpos);
    sim_kernel<<<dim3(B_, MAXSIMS), 256, 0, stream>>>(
        hidden, rmean, fpos, sims, Wt1, bt1, Wt2, bt2, Wav1, bav1, Wav2, bav2, accb, wsum);
    acc_finalize<<<(int)(BSH / 256), 256, 0, stream>>>(hidden, wsum, accb);

    // aggregation (WMMA): h1 = GELU([root|acc] @ Wg1 + bg1); out = hidden + h1 @ Wg2 + bg2
    gemm_wmma_bf16<1><<<dim3(H_ / BN, (B_ * S_) / BM), 256, 0, stream>>>(
        hidden, accb, H_, Wg1, bg1, nullptr, h1, B_ * S_, H_, 2 * H_);
    gemm_wmma_bf16<2><<<dim3(H_ / BN, (B_ * S_) / BM), 256, 0, stream>>>(
        h1, nullptr, H_, Wg2, bg2, hidden, out, B_ * S_, H_, H_);
}
